// GroupAttention_86784109183341
// MI455X (gfx1250) — matrix operand order repaired and checked by reading, not compiled, not run on hardware
//
#include <hip/hip_runtime.h>
#include <hip/hip_bf16.h>

// ---------------------------------------------------------------------------
// Transformer-XL style relative attention block for MI455X (gfx1250, wave32).
// All big matmuls run on v_wmma_f32_16x16x32_bf16 (fp32 accumulate).
// ---------------------------------------------------------------------------

#define QL     512
#define BATCH  4
#define KLENC  1024
#define DMOD   1024
#define NHEAD  16
#define DH     64
#define KDIM   1024
#define SCALE_C 0.125f        // 1/sqrt(64)
#define EPS_C   1e-6f

typedef __attribute__((ext_vector_type(16))) __bf16 bf16x16;
typedef __attribute__((ext_vector_type(8)))  __bf16 bf16x8;
typedef __attribute__((ext_vector_type(4)))  __bf16 bf16x4;
typedef __attribute__((ext_vector_type(8)))  float  f32x8;
typedef __attribute__((ext_vector_type(4)))  float  f32x4;

// ---- helpers --------------------------------------------------------------

__device__ __forceinline__ __bf16 f2bf(float f) {
  unsigned u = __builtin_bit_cast(unsigned, f);
  unsigned r = u + 0x7FFFu + ((u >> 16) & 1u);   // round-to-nearest-even
  unsigned short s = (unsigned short)(r >> 16);
  return __builtin_bit_cast(__bf16, s);
}

__device__ __forceinline__ float bf2f(__bf16 x) {
  unsigned short u = __builtin_bit_cast(unsigned short, x);
  unsigned v = ((unsigned)u) << 16;
  return __builtin_bit_cast(float, v);
}

__device__ __forceinline__ bf16x16 mk16(bf16x8 lo, bf16x8 hi) {
  bf16x16 r;
#pragma unroll
  for (int i = 0; i < 8; i++) { r[i] = lo[i]; r[i + 8] = hi[i]; }
  return r;
}

__device__ __forceinline__ f32x8 wmma_bf16(bf16x16 a, bf16x16 b, f32x8 c) {
  return __builtin_amdgcn_wmma_f32_16x16x32_bf16(false, a, false, b,
                                                 (short)0, c, false, false);
}

// ---------------------------------------------------------------------------
// Kernel 1: pack weights.
//   Wqc[o][d] = Wiq[o%256][d] + (same group ? Wq[g][o%256][d%256] : 0)
//   Wrc[o][d] = same group ? Wr[g][o%256][d%256] : 0
//   Woc[o][d] = Winter[o%256][d] + (same group ? Wintra[g][o%256][d%256] : 0)
//   plus straight bf16 conversions of Wk, Wv and r.
// ---------------------------------------------------------------------------
__global__ __launch_bounds__(256) void pack_weights_kernel(
    const float* __restrict__ Wq, const float* __restrict__ Wiq,
    const float* __restrict__ Wk, const float* __restrict__ Wv,
    const float* __restrict__ Wr, const float* __restrict__ Wintra,
    const float* __restrict__ Winter, const float* __restrict__ rsrc,
    __bf16* __restrict__ Wqc, __bf16* __restrict__ Wrc,
    __bf16* __restrict__ Woc, __bf16* __restrict__ Wkb,
    __bf16* __restrict__ Wvb, __bf16* __restrict__ rbf)
{
  int idx = blockIdx.x * 256 + threadIdx.x;      // 0 .. 1024*1024-1
  int o = idx >> 10, d = idx & 1023;
  int g = o >> 8, og = o & 255, dg = d & 255;
  bool same = ((d >> 8) == g);
  int gidx = (g * 256 + og) * 256 + dg;

  float wq = Wiq[og * 1024 + d] + (same ? Wq[gidx] : 0.0f);
  float wr = same ? Wr[gidx] : 0.0f;
  float wo = Winter[og * 1024 + d] + (same ? Wintra[gidx] : 0.0f);

  Wqc[idx] = f2bf(wq);
  Wrc[idx] = f2bf(wr);
  Woc[idx] = f2bf(wo);
  Wkb[idx] = f2bf(Wk[idx]);
  Wvb[idx] = f2bf(Wv[idx]);
  rbf[idx] = f2bf(rsrc[idx]);   // r is [1024,1,1024] -> flat 1M
}

// ---------------------------------------------------------------------------
// Kernel 2: group LayerNorm of w (G=4 groups of 256), unbiased std (ddof=1),
//           out = gamma*(x-mean)/(std+eps)+beta, stored bf16.
// One block (4 waves) per row; wave g handles group g; 8 elems/lane.
// ---------------------------------------------------------------------------
__global__ __launch_bounds__(128) void groupln_w_kernel(
    const float* __restrict__ w, const float* __restrict__ gamma,
    const float* __restrict__ beta, __bf16* __restrict__ out)
{
  int row = blockIdx.x;                       // 0..2047  (= q*B + b)
  int wid = threadIdx.x >> 5, lane = threadIdx.x & 31;
  int cbase = wid * 256 + lane * 8;
  const float* x = w + (size_t)row * DMOD + cbase;

  float v[8];
  f32x4 a = *(const f32x4*)(x);
  f32x4 c = *(const f32x4*)(x + 4);
#pragma unroll
  for (int i = 0; i < 4; i++) { v[i] = a[i]; v[i + 4] = c[i]; }

  float s = 0.f, sq = 0.f;
#pragma unroll
  for (int i = 0; i < 8; i++) { s += v[i]; sq += v[i] * v[i]; }
#pragma unroll
  for (int m = 1; m < 32; m <<= 1) {
    s  += __shfl_xor(s, m, 32);
    sq += __shfl_xor(sq, m, 32);
  }
  float mean = s * (1.0f / 256.0f);
  float var  = (sq - s * mean) * (1.0f / 255.0f);
  var = fmaxf(var, 0.0f);
  float denom = sqrtf(var) + EPS_C;

  bf16x8 ob;
#pragma unroll
  for (int i = 0; i < 8; i++) {
    float y = gamma[cbase + i] * ((v[i] - mean) / denom) + beta[cbase + i];
    ob[i] = f2bf(y);
  }
  *(bf16x8*)(out + (size_t)row * DMOD + cbase) = ob;
}

// ---------------------------------------------------------------------------
// Kernel 3: full-row LayerNorm of kv = concat(mems, w), rows = k*B+b (4096).
// ---------------------------------------------------------------------------
__global__ __launch_bounds__(256) void ln_kv_kernel(
    const float* __restrict__ mems, const float* __restrict__ w,
    const float* __restrict__ gamma, const float* __restrict__ beta,
    __bf16* __restrict__ out)
{
  __shared__ float ps[8], pq[8];
  int row = blockIdx.x;                       // 0..4095
  const float* src = (row < 2048) ? mems + (size_t)row * DMOD
                                  : w + (size_t)(row - 2048) * DMOD;
  int tid = threadIdx.x;
  int cbase = tid * 4;
  f32x4 v = *(const f32x4*)(src + cbase);
  float s = v[0] + v[1] + v[2] + v[3];
  float sq = v[0]*v[0] + v[1]*v[1] + v[2]*v[2] + v[3]*v[3];
#pragma unroll
  for (int m = 1; m < 32; m <<= 1) {
    s  += __shfl_xor(s, m, 32);
    sq += __shfl_xor(sq, m, 32);
  }
  int wid = tid >> 5;
  if ((tid & 31) == 0) { ps[wid] = s; pq[wid] = sq; }
  __syncthreads();
  float S = 0.f, SQ = 0.f;
#pragma unroll
  for (int k = 0; k < 8; k++) { S += ps[k]; SQ += pq[k]; }
  float mean = S * (1.0f / 1024.0f);
  float var  = (SQ - S * mean) * (1.0f / 1023.0f);
  var = fmaxf(var, 0.0f);
  float denom = sqrtf(var) + EPS_C;

  bf16x4 ob;
#pragma unroll
  for (int i = 0; i < 4; i++) {
    float y = gamma[cbase + i] * ((v[i] - mean) / denom) + beta[cbase + i];
    ob[i] = f2bf(y);
  }
  *(bf16x4*)(out + (size_t)row * DMOD + cbase) = ob;
}

// ---------------------------------------------------------------------------
// Generic WMMA GEMM core: C[16x64 per wave] = A[Mx1024] @ Bt[1024x1024]^T.
// A row-major [M][K]; Bt row-major [N][K] (i.e. the weight as stored).
// A and B fragments use contiguous k = 16*half + s (16-bit A 16x32, B 32x16).
// ---------------------------------------------------------------------------
template <class Epi>
__device__ __forceinline__ void gemm_wave(
    const __bf16* __restrict__ A, const __bf16* __restrict__ Bt,
    int m0, int n0, Epi epi)
{
  const int lane = threadIdx.x & 31;
  const int half = lane >> 4, l16 = lane & 15;

  f32x8 acc[4] = {};
  const __bf16* arow = A + (size_t)(m0 + l16) * KDIM;
  const __bf16* bbase = Bt + (size_t)(n0 + l16) * KDIM + 16 * half;

#pragma unroll 2
  for (int k0 = 0; k0 < KDIM; k0 += 32) {
    bf16x16 a = mk16(*(const bf16x8*)(arow + k0 + 16 * half),
                     *(const bf16x8*)(arow + k0 + 16 * half + 8));
#pragma unroll
    for (int t = 0; t < 4; t++) {
      const __bf16* bp = bbase + (size_t)t * 16 * KDIM + k0;
      bf16x16 bb = mk16(*(const bf16x8*)(bp), *(const bf16x8*)(bp + 8));
      acc[t] = wmma_bf16(a, bb, acc[t]);
    }
  }
#pragma unroll
  for (int t = 0; t < 4; t++) epi(t, acc[t], half, l16);
}

// head_q projection: writes two biased bf16 copies (q+r_w_bias, q+r_r_bias).
__global__ __launch_bounds__(128) void gemm_q_kernel(
    const __bf16* __restrict__ A, const __bf16* __restrict__ Bt,
    const float* __restrict__ rwb, const float* __restrict__ rrb,
    __bf16* __restrict__ qw, __bf16* __restrict__ qr)
{
  int m0 = (blockIdx.x * 4 + (threadIdx.x >> 5)) * 16;
  int n0 = blockIdx.y * 64;
  gemm_wave(A, Bt, m0, n0, [&](int t, const f32x8& c, int half, int l16) {
    int n = n0 + t * 16 + l16;
    float bw = rwb[n], br = rrb[n];
#pragma unroll
    for (int v = 0; v < 8; v++) {
      size_t row = (size_t)(m0 + v + 8 * half);
      float x = c[v];
      qw[row * DMOD + n] = f2bf(x + bw);
      qr[row * DMOD + n] = f2bf(x + br);
    }
  });
}

// plain projection (head_k, r_head_k): C row-major bf16.
__global__ __launch_bounds__(128) void gemm_plain_kernel(
    const __bf16* __restrict__ A, const __bf16* __restrict__ Bt,
    __bf16* __restrict__ C)
{
  int m0 = (blockIdx.x * 4 + (threadIdx.x >> 5)) * 16;
  int n0 = blockIdx.y * 64;
  gemm_wave(A, Bt, m0, n0, [&](int t, const f32x8& c, int half, int l16) {
    int n = n0 + t * 16 + l16;
#pragma unroll
    for (int v = 0; v < 8; v++) {
      size_t row = (size_t)(m0 + v + 8 * half);
      C[row * DMOD + n] = f2bf(c[v]);
    }
  });
}

// head_v projection with transposed store: vT[b][h][d][k] (k contiguous).
__global__ __launch_bounds__(128) void gemm_vT_kernel(
    const __bf16* __restrict__ A, const __bf16* __restrict__ Bt,
    __bf16* __restrict__ vT)
{
  int m0 = (blockIdx.x * 4 + (threadIdx.x >> 5)) * 16;
  int n0 = blockIdx.y * 64;
  gemm_wave(A, Bt, m0, n0, [&](int t, const f32x8& c, int half, int l16) {
    int n = n0 + t * 16 + l16;
    int h = n >> 6, d = n & 63;
#pragma unroll
    for (int v = 0; v < 8; v++) {
      int rowi = m0 + v + 8 * half;        // = k*B + b
      int k = rowi >> 2, b = rowi & 3;
      vT[((size_t)(b * NHEAD + h) * DH + d) * KLENC + k] = f2bf(c[v]);
    }
  });
}

// output projection + residual: out = w + attn_out @ Woc^T  (fp32 out).
__global__ __launch_bounds__(128) void gemm_out_kernel(
    const __bf16* __restrict__ A, const __bf16* __restrict__ Bt,
    const float* __restrict__ wres, float* __restrict__ out)
{
  int m0 = (blockIdx.x * 4 + (threadIdx.x >> 5)) * 16;
  int n0 = blockIdx.y * 64;
  gemm_wave(A, Bt, m0, n0, [&](int t, const f32x8& c, int half, int l16) {
    int n = n0 + t * 16 + l16;
#pragma unroll
    for (int v = 0; v < 8; v++) {
      size_t row = (size_t)(m0 + v + 8 * half);
      out[row * DMOD + n] = wres[row * DMOD + n] + c[v];
    }
  });
}

// ---------------------------------------------------------------------------
// Kernel: fused relative attention per (b, h, 16-row q-tile).
//  Phase 1: bd_raw[16x1024] = (q+r_r_bias) @ r_head_k^T  -> LDS (bf16, WMMA)
//  Phase 2: flash loop over j in steps of 32:
//           S = (q+r_w_bias)@K^T (WMMA) + shifted bd_raw read from LDS,
//           mask j>i+512, online softmax, P@V via WMMA (V pre-transposed).
//  rel_shift: BD[i,j] = bd_raw[i, j+511-i]; wrapped values land on masked
//  positions only, so the analytic mask covers them.
// ---------------------------------------------------------------------------
#define BDS 1032   // padded LDS row stride for bd (bank-conflict avoidance)
#define PST 40     // padded LDS row stride for P staging

__global__ __launch_bounds__(32) void attn_kernel(
    const __bf16* __restrict__ qw, const __bf16* __restrict__ qr,
    const __bf16* __restrict__ hk, const __bf16* __restrict__ vT,
    const __bf16* __restrict__ rk, __bf16* __restrict__ aout)
{
  __shared__ __attribute__((aligned(16))) __bf16 bd[16 * BDS];
  __shared__ __attribute__((aligned(16))) __bf16 pl[16 * PST];

  const int it = blockIdx.x;            // q tile: 0..31
  const int bh = blockIdx.y;            // b*16 + h: 0..63
  const int b = bh >> 4, h = bh & 15;
  const int i0 = it * 16;
  const int lane = threadIdx.x & 31;
  const int half = lane >> 4, l16 = lane & 15;

  // --- load q fragments (16 rows x 64 K, as two 16x32 A fragments each) ---
  const __bf16* qwrow = qw + ((size_t)(i0 + l16) * BATCH + b) * DMOD + h * DH;
  const __bf16* qrrow = qr + ((size_t)(i0 + l16) * BATCH + b) * DMOD + h * DH;
  bf16x16 qwA[2], qrA[2];
#pragma unroll
  for (int ks = 0; ks < 2; ks++) {
    qwA[ks] = mk16(*(const bf16x8*)(qwrow + ks * 32 + 16 * half),
                   *(const bf16x8*)(qwrow + ks * 32 + 16 * half + 8));
    qrA[ks] = mk16(*(const bf16x8*)(qrrow + ks * 32 + 16 * half),
                   *(const bf16x8*)(qrrow + ks * 32 + 16 * half + 8));
  }

  // --- Phase 1: bd_raw -> LDS (M=16, N=1024, K=64) ---
#pragma unroll 1
  for (int nt = 0; nt < 64; nt++) {
    f32x8 acc = {};
    const __bf16* rrow = rk + (size_t)(nt * 16 + l16) * DMOD + h * DH + 16 * half;
#pragma unroll
    for (int ks = 0; ks < 2; ks++) {
      bf16x16 bb = mk16(*(const bf16x8*)(rrow + ks * 32),
                        *(const bf16x8*)(rrow + ks * 32 + 8));
      acc = wmma_bf16(qrA[ks], bb, acc);
    }
#pragma unroll
    for (int v = 0; v < 8; v++)
      bd[(v + 8 * half) * BDS + nt * 16 + l16] = f2bf(acc[v]);
  }
  __syncthreads();

  // --- Phase 2: flash attention over j ---
  f32x8 o[4] = {};
  float mi[8], li[8];
#pragma unroll
  for (int v = 0; v < 8; v++) { mi[v] = -__builtin_inff(); li[v] = 0.0f; }

  int jhi = (i0 + 527) / 32 + 1;
  if (jhi > 32) jhi = 32;

#pragma unroll 1
  for (int jt = 0; jt < jhi; jt++) {
    const int j0 = jt * 32;
    f32x8 s[2] = {};
#pragma unroll
    for (int a = 0; a < 2; a++) {
      const __bf16* krow =
          hk + ((size_t)(j0 + a * 16 + l16) * BATCH + b) * DMOD + h * DH + 16 * half;
#pragma unroll
      for (int ks = 0; ks < 2; ks++) {
        bf16x16 bb = mk16(*(const bf16x8*)(krow + ks * 32),
                          *(const bf16x8*)(krow + ks * 32 + 8));
        s[a] = wmma_bf16(qwA[ks], bb, s[a]);
      }
    }

#pragma unroll
    for (int v = 0; v < 8; v++) {
      const int m = v + 8 * half;
      const int i = i0 + m;
      float sv[2];
#pragma unroll
      for (int a = 0; a < 2; a++) {
        int j = j0 + a * 16 + l16;
        int r = j + 511 - i;               // rel_shift gather index
        int rr = r < 1023 ? r : 1023;
        float x = (s[a][v] + bf2f(bd[m * BDS + rr])) * SCALE_C;
        if (j > i + 512) x = -1e30f;       // causal+mem mask
        sv[a] = x;
      }
      float tm = fmaxf(sv[0], sv[1]);
      tm = fmaxf(tm, __shfl_xor(tm, 1, 32));
      tm = fmaxf(tm, __shfl_xor(tm, 2, 32));
      tm = fmaxf(tm, __shfl_xor(tm, 4, 32));
      tm = fmaxf(tm, __shfl_xor(tm, 8, 32));
      float mn = fmaxf(mi[v], tm);
      float al = __expf(mi[v] - mn);
      float p0 = __expf(sv[0] - mn);
      float p1 = __expf(sv[1] - mn);
      float psum = p0 + p1;
      psum += __shfl_xor(psum, 1, 32);
      psum += __shfl_xor(psum, 2, 32);
      psum += __shfl_xor(psum, 4, 32);
      psum += __shfl_xor(psum, 8, 32);
      li[v] = li[v] * al + psum;
      mi[v] = mn;
#pragma unroll
      for (int t = 0; t < 4; t++) o[t][v] = o[t][v] * al;
      pl[m * PST + l16]      = f2bf(p0);
      pl[m * PST + 16 + l16] = f2bf(p1);
    }
    __syncthreads();

    // P as a 16x32 A fragment from LDS
    bf16x16 pa = mk16(*(const bf16x8*)(pl + l16 * PST + 16 * half),
                      *(const bf16x8*)(pl + l16 * PST + 16 * half + 8));
#pragma unroll
    for (int t = 0; t < 4; t++) {
      const __bf16* vrow =
          vT + ((size_t)bh * DH + t * 16 + l16) * KLENC + j0 + 16 * half;
      bf16x16 bb = mk16(*(const bf16x8*)(vrow), *(const bf16x8*)(vrow + 8));
      o[t] = wmma_bf16(pa, bb, o[t]);
    }
    __syncthreads();
  }

  // --- epilogue: normalize and store attn_out (bf16, natural layout) ---
#pragma unroll
  for (int v = 0; v < 8; v++) {
    const int i = i0 + v + 8 * half;
    float inv = 1.0f / li[v];
#pragma unroll
    for (int t = 0; t < 4; t++) {
      aout[((size_t)i * BATCH + b) * DMOD + h * DH + t * 16 + l16] =
          f2bf(o[t][v] * inv);
    }
  }
}

// ---------------------------------------------------------------------------
// Host launcher
// ---------------------------------------------------------------------------
extern "C" void kernel_launch(void* const* d_in, const int* in_sizes, int n_in,
                              void* d_out, int out_size, void* d_ws, size_t ws_size,
                              hipStream_t stream)
{
  (void)in_sizes; (void)n_in; (void)out_size; (void)ws_size;

  const float* w      = (const float*)d_in[0];
  const float* r      = (const float*)d_in[1];
  const float* rwb    = (const float*)d_in[2];
  const float* rrb    = (const float*)d_in[3];
  const float* mems   = (const float*)d_in[4];
  const float* gq     = (const float*)d_in[5];
  const float* bq     = (const float*)d_in[6];
  const float* gkv    = (const float*)d_in[7];
  const float* bkv    = (const float*)d_in[8];
  const float* Wq     = (const float*)d_in[9];
  const float* Wiq    = (const float*)d_in[10];
  const float* Wk     = (const float*)d_in[11];
  const float* Wv     = (const float*)d_in[12];
  const float* Wr     = (const float*)d_in[13];
  const float* Wintra = (const float*)d_in[14];
  const float* Winter = (const float*)d_in[15];
  // d_in[16] = attn_mask: computed analytically (j > i + 512), unused.

  float* out = (float*)d_out;

  char* p = (char*)d_ws;
  auto carve = [&](size_t bytes) -> char* {
    char* q = p; p += (bytes + 255) & ~(size_t)255; return q;
  };
  const size_t MB2 = (size_t)1024 * 1024 * 2;  // 1M bf16

  __bf16* Wqc   = (__bf16*)carve(MB2);
  __bf16* Wrc   = (__bf16*)carve(MB2);
  __bf16* Woc   = (__bf16*)carve(MB2);
  __bf16* Wkb   = (__bf16*)carve(MB2);
  __bf16* Wvb   = (__bf16*)carve(MB2);
  __bf16* rbf   = (__bf16*)carve(MB2);
  __bf16* wnorm = (__bf16*)carve((size_t)2048 * 1024 * 2);
  __bf16* kvn   = (__bf16*)carve((size_t)4096 * 1024 * 2);
  __bf16* qwb   = (__bf16*)carve((size_t)2048 * 1024 * 2);
  __bf16* qrb   = (__bf16*)carve((size_t)2048 * 1024 * 2);
  __bf16* hk    = (__bf16*)carve((size_t)4096 * 1024 * 2);
  __bf16* vT    = (__bf16*)carve((size_t)4096 * 1024 * 2);
  __bf16* rk    = (__bf16*)carve((size_t)1024 * 1024 * 2);
  __bf16* aoutb = (__bf16*)carve((size_t)2048 * 1024 * 2);

  // 1. pack weights + convert r
  pack_weights_kernel<<<4096, 256, 0, stream>>>(
      Wq, Wiq, Wk, Wv, Wr, Wintra, Winter, r,
      Wqc, Wrc, Woc, Wkb, Wvb, rbf);

  // 2. layernorms
  groupln_w_kernel<<<2048, 128, 0, stream>>>(w, gq, bq, wnorm);
  ln_kv_kernel<<<4096, 256, 0, stream>>>(mems, w, gkv, bkv, kvn);

  // 3. projections (all WMMA GEMMs, K=1024, N=1024)
  gemm_q_kernel<<<dim3(32, 16), 128, 0, stream>>>(wnorm, Wqc, rwb, rrb, qwb, qrb);
  gemm_plain_kernel<<<dim3(64, 16), 128, 0, stream>>>(kvn, Wkb, hk);
  gemm_vT_kernel<<<dim3(64, 16), 128, 0, stream>>>(kvn, Wvb, vT);
  gemm_plain_kernel<<<dim3(16, 16), 128, 0, stream>>>(rbf, Wrc, rk);

  // 4. fused relative attention (one wave per (b,h,q-tile))
  attn_kernel<<<dim3(32, 64), 32, 0, stream>>>(qwb, qrb, hk, vT, rk, aoutb);

  // 5. output projection + residual
  gemm_out_kernel<<<dim3(32, 16), 128, 0, stream>>>(aoutb, Woc, w, out);
}
